// Transformer_3083786518928
// MI455X (gfx1250) — matrix operand order repaired and checked by reading, not compiled, not run on hardware
//
#include <hip/hip_runtime.h>

typedef __attribute__((ext_vector_type(16))) __bf16 v16bf;
typedef __attribute__((ext_vector_type(8)))  __bf16 v8bf;
typedef __attribute__((ext_vector_type(8)))  float  v8f;

namespace {
constexpr int kD  = 512;     // d_model
constexpr int kFF = 2048;    // d_ff
constexpr int kS  = 1024;    // seq len (== T)
constexpr int kB  = 2;       // batch
constexpr int kH  = 8;       // heads
constexpr int kHD = 64;      // head dim
constexpr int kL  = 4;       // layers
constexpr int kV  = 32000;   // vocab
constexpr int kNT = kB * kS; // total tokens (2048)
}

// ---------------- device helpers ----------------

__device__ __forceinline__ __bf16 f2bf(float f) {
  unsigned int u = __builtin_bit_cast(unsigned int, f);
  u += 0x7FFFu + ((u >> 16) & 1u);                 // round-to-nearest-even
  unsigned short h = (unsigned short)(u >> 16);
  return __builtin_bit_cast(__bf16, h);
}

__device__ __forceinline__ v16bf load_afrag(const __bf16* arow, int k0) {
  // ISA 16-bit A layout per lane: K chunks {8g..8g+7} and {16+8g..16+8g+7}
  // (g already folded into arow). Two contiguous 16B loads.
  const v8bf a0 = *(const v8bf*)(arow + k0);
  const v8bf a1 = *(const v8bf*)(arow + k0 + 16);
  return __builtin_shufflevector(a0, a1, 0, 1, 2, 3, 4, 5, 6, 7,
                                 8, 9, 10, 11, 12, 13, 14, 15);
}

// ---------------- WMMA bf16 GEMM, B pre-transposed ----------------
// C = scale * A(MxK) * B(KxN) [+bias] [relu] [+Cin(res)] -> Cout f32 / Cbout bf16.
// B stored transposed: Bt[n*ldb + k]  => lane fragment = same two 16B loads as A.
// Batched over blockIdx.z: (bb,hh) = (z/HH, z%HH) with per-operand strides.
// Block: 256 threads = 8 waves. Wave tile 32x64 (2 A frags x 4 B frags = 8 WMMA
// per K-step, 12 16B loads : 8 WMMAs). Block tile 256x64.
// Requires M%256==0, N%64==0, K%32==0 (true for all shapes used here).
template <bool BIAS, bool RES, bool RELU, bool WF32, bool WB16>
__global__ void __launch_bounds__(256)
gemm_wmma(const __bf16* __restrict__ A, long lda,
          const __bf16* __restrict__ Bt, long ldb,
          const float* __restrict__ bias,
          const float* __restrict__ Cin,
          float* __restrict__ Cout,
          __bf16* __restrict__ Cbout,
          long ldc, int M, int N, int K, float scale,
          int HH, long sAb, long sAh, long sBb, long sBh,
          long sCb, long sCh)
{
  const int z  = blockIdx.z;
  const int bb = z / HH;
  const int hh = z - bb * HH;
  A  += (long)bb * sAb + (long)hh * sAh;
  Bt += (long)bb * sBb + (long)hh * sBh;
  const long coff = (long)bb * sCb + (long)hh * sCh;

  const int lane = threadIdx.x & 31;
  const int wave = threadIdx.x >> 5;
  const int g    = lane >> 4;     // lane group
  const int ln   = lane & 15;
  const int tileM = blockIdx.y * 256 + wave * 32;
  const int tileN = blockIdx.x * 64;

  v8f acc0[4], acc1[4];
#pragma unroll
  for (int t = 0; t < 4; ++t)
#pragma unroll
    for (int r = 0; r < 8; ++r) { acc0[t][r] = 0.0f; acc1[t][r] = 0.0f; }

  const __bf16* arow0 = A + (long)(tileM + ln) * lda + g * 8;       // rows m..m+15
  const __bf16* arow1 = arow0 + 16 * lda;                            // rows m+16..m+31
  const __bf16* brow  = Bt + (long)(tileN + ln) * ldb + g * 8;
  const long bstep = 16 * ldb;

  for (int k0 = 0; k0 < K; k0 += 32) {
    const v16bf af0 = load_afrag(arow0, k0);
    const v16bf af1 = load_afrag(arow1, k0);
#pragma unroll
    for (int t = 0; t < 4; ++t) {
      const v16bf bfrag = load_afrag(brow + (long)t * bstep, k0);
      acc0[t] = __builtin_amdgcn_wmma_f32_16x16x32_bf16(
          false, af0, false, bfrag, (short)0, acc0[t], false, false);
      acc1[t] = __builtin_amdgcn_wmma_f32_16x16x32_bf16(
          false, af1, false, bfrag, (short)0, acc1[t], false, false);
    }
  }

  // C/D layout: VGPR r -> row = rowBase + 8g + r, col = tileN + 16t + ln
#pragma unroll
  for (int t = 0; t < 4; ++t) {
    const int col = tileN + t * 16 + ln;
    float bv = 0.0f;
    if constexpr (BIAS) bv = bias[col];
#pragma unroll
    for (int half = 0; half < 2; ++half) {
      const v8f& a = half ? acc1[t] : acc0[t];
      const int rowBase = tileM + half * 16 + g * 8;
#pragma unroll
      for (int r = 0; r < 8; ++r) {
        float v = a[r] * scale;
        if constexpr (BIAS) v += bv;
        if constexpr (RELU) v = fmaxf(v, 0.0f);
        const long o = coff + (long)(rowBase + r) * ldc + col;
        if constexpr (RES)  v += Cin[o];
        if constexpr (WF32) Cout[o] = v;
        if constexpr (WB16) Cbout[o] = f2bf(v);
      }
    }
  }
}

// ---------------- LayerNorm (torch unbiased var), writes bf16 ----------------
__global__ void __launch_bounds__(256)
layernorm_bf16(const float* __restrict__ X, const float* __restrict__ gamma,
               const float* __restrict__ beta, __bf16* __restrict__ Z, int nrows)
{
  const int row = blockIdx.x * 8 + (threadIdx.x >> 5);
  if (row >= nrows) return;
  const int lane = threadIdx.x & 31;
  const float* x = X + (long)row * kD;

  float v[kD / 32];
  float s = 0.0f;
#pragma unroll
  for (int j = 0; j < kD / 32; ++j) { v[j] = x[lane + j * 32]; s += v[j]; }
  for (int off = 16; off; off >>= 1) s += __shfl_xor(s, off, 32);
  const float mean = s * (1.0f / kD);
  float q = 0.0f;
#pragma unroll
  for (int j = 0; j < kD / 32; ++j) { const float d = v[j] - mean; q += d * d; }
  for (int off = 16; off; off >>= 1) q += __shfl_xor(q, off, 32);
  const float rstd = rsqrtf(q * (1.0f / (kD - 1)) + 1e-6f);

  __bf16* zr = Z + (long)row * kD;
#pragma unroll
  for (int j = 0; j < kD / 32; ++j) {
    const int d = lane + j * 32;
    zr[d] = f2bf(gamma[d] * ((v[j] - mean) * rstd) + beta[d]);
  }
}

// ---------------- row softmax over 1024 cols, optional causal, bf16 out ----------------
__global__ void __launch_bounds__(256)
softmax1024(const float* __restrict__ X, __bf16* __restrict__ P,
            int causal, int nrows)
{
  const int row = blockIdx.x * 8 + (threadIdx.x >> 5);
  if (row >= nrows) return;
  const int lane = threadIdx.x & 31;
  const int qpos = row & (kS - 1);
  const float* xr = X + (long)row * kS;

  float v[kS / 32];
  float m = -__builtin_inff();
#pragma unroll
  for (int t = 0; t < kS / 32; ++t) {
    const int j = lane + t * 32;
    float x = xr[j];
    if (causal && j > qpos) x = -__builtin_inff();
    v[t] = x;
    m = fmaxf(m, x);
  }
  for (int off = 16; off; off >>= 1) m = fmaxf(m, __shfl_xor(m, off, 32));
  float s = 0.0f;
#pragma unroll
  for (int t = 0; t < kS / 32; ++t) { v[t] = expf(v[t] - m); s += v[t]; }
  for (int off = 16; off; off >>= 1) s += __shfl_xor(s, off, 32);
  const float inv = 1.0f / s;

  __bf16* pr = P + (long)row * kS;
#pragma unroll
  for (int t = 0; t < kS / 32; ++t) pr[lane + t * 32] = f2bf(v[t] * inv);
}

// ---------------- token embedding * sqrt(D) + sinusoidal posenc ----------------
__global__ void __launch_bounds__(256)
embed_pe(const float* __restrict__ emb, const int* __restrict__ tok,
         float* __restrict__ X, int nrows)
{
  const long i = (long)blockIdx.x * blockDim.x + threadIdx.x;
  if (i >= (long)nrows * kD) return;
  const int d = (int)(i & (kD - 1));
  const long row = i >> 9;
  const int s = (int)(row & (kS - 1));
  const int t = tok[row];
  const float dv = expf(-(float)(d & ~1) * (logf(10000.0f) / (float)kD));
  const float ang = (float)s * dv;
  const float pe = (d & 1) ? cosf(ang) : sinf(ang);
  X[i] = emb[(long)t * kD + d] * 22.62741699796952f + pe;
}

// ---------------- fused f32 -> bf16 convert + transpose ----------------
// in:  [nb, K, N] f32 ; out: [nb, N, K] bf16
__global__ void __launch_bounds__(256)
cvt_transpose(const float* __restrict__ in, __bf16* __restrict__ o,
              int K, int N, long total)
{
  const long i = (long)blockIdx.x * 256 + threadIdx.x;
  if (i >= total) return;
  const long k = i % K;
  const long r = i / K;
  const long n = r % N;
  const long b = r / N;
  o[i] = f2bf(in[(b * K + k) * N + n]);
}

// ---------------- per-head transpose of V: [B*S,512] -> [B*H, 64, S] bf16 ----------------
__global__ void __launch_bounds__(256)
transpose_v(const __bf16* __restrict__ V, __bf16* __restrict__ vT)
{
  const long i = (long)blockIdx.x * 256 + threadIdx.x;  // exactly B*H*64*S threads
  const int  k  = (int)(i & (kS - 1));
  const long r  = i >> 10;
  const int  n  = (int)(r & (kHD - 1));
  const long zh = r >> 6;                 // b*H + h
  const long b  = zh >> 3;
  const int  h  = (int)(zh & (kH - 1));
  vT[i] = V[(b * kS + k) * kD + (long)h * kHD + n];
}

// ---------------- host orchestration ----------------

extern "C" void kernel_launch(void* const* d_in, const int* in_sizes, int n_in,
                              void* d_out, int out_size, void* d_ws, size_t ws_size,
                              hipStream_t stream) {
  (void)in_sizes; (void)n_in; (void)out_size; (void)ws_size;

  const int*   src        = (const int*)d_in[0];
  const int*   tgt        = (const int*)d_in[1];
  const float* src_emb    = (const float*)d_in[4];
  const float* tgt_emb    = (const float*)d_in[5];
  const float* enc_att_w  = (const float*)d_in[6];
  const float* enc_ff_w1  = (const float*)d_in[7];
  const float* enc_ff_b1  = (const float*)d_in[8];
  const float* enc_ff_w2  = (const float*)d_in[9];
  const float* enc_ff_b2  = (const float*)d_in[10];
  const float* enc_norm_g = (const float*)d_in[11];
  const float* enc_norm_b = (const float*)d_in[12];
  const float* enc_fin_g  = (const float*)d_in[13];
  const float* enc_fin_b  = (const float*)d_in[14];
  const float* dec_att_w  = (const float*)d_in[15];
  const float* dec_ff_w1  = (const float*)d_in[16];
  const float* dec_ff_b1  = (const float*)d_in[17];
  const float* dec_ff_w2  = (const float*)d_in[18];
  const float* dec_ff_b2  = (const float*)d_in[19];
  const float* dec_norm_g = (const float*)d_in[20];
  const float* dec_norm_b = (const float*)d_in[21];
  const float* dec_fin_g  = (const float*)d_in[22];
  const float* dec_fin_b  = (const float*)d_in[23];
  const float* proj_w     = (const float*)d_in[24];
  const float* proj_b     = (const float*)d_in[25];
  float* out = (float*)d_out;

  // ---- workspace carve (all 256B aligned) ----
  char* wp = (char*)d_ws;
  auto carve = [&](size_t bytes) -> void* {
    void* r = (void*)wp;
    wp += (bytes + 255) & ~(size_t)255;
    return r;
  };
  // all weights stored bf16 TRANSPOSED: [N, K]
  __bf16* w_enc_att = (__bf16*)carve((size_t)kL * 4 * kD * kD * 2);
  __bf16* w_enc_f1  = (__bf16*)carve((size_t)kL * kFF * kD * 2);
  __bf16* w_enc_f2  = (__bf16*)carve((size_t)kL * kD * kFF * 2);
  __bf16* w_dec_att = (__bf16*)carve((size_t)kL * 2 * 4 * kD * kD * 2);
  __bf16* w_dec_f1  = (__bf16*)carve((size_t)kL * kFF * kD * 2);
  __bf16* w_dec_f2  = (__bf16*)carve((size_t)kL * kD * kFF * 2);
  __bf16* w_proj    = (__bf16*)carve((size_t)kV * kD * 2);
  float*  x_enc     = (float*)carve((size_t)kNT * kD * 4);
  float*  x_dec     = (float*)carve((size_t)kNT * kD * 4);
  __bf16* zb        = (__bf16*)carve((size_t)kNT * kD * 2);
  __bf16* qb        = (__bf16*)carve((size_t)kNT * kD * 2);
  __bf16* kbuf      = (__bf16*)carve((size_t)kNT * kD * 2);
  __bf16* vbuf      = (__bf16*)carve((size_t)kNT * kD * 2);
  __bf16* vT        = (__bf16*)carve((size_t)kB * kH * kHD * kS * 2);
  __bf16* enc_b     = (__bf16*)carve((size_t)kNT * kD * 2);
  __bf16* ctx       = (__bf16*)carve((size_t)kNT * kD * 2);
  __bf16* ffh       = (__bf16*)carve((size_t)kNT * kFF * 2);
  float*  scores    = (float*)carve((size_t)kB * kH * kS * kS * 4);
  __bf16* probs     = (__bf16*)carve((size_t)kB * kH * kS * kS * 2);

  // ---- launch helpers ----
  auto cvtT = [&](const float* in, __bf16* o, int nb, int K, int N) {
    const long total = (long)nb * K * N;
    cvt_transpose<<<dim3((unsigned)((total + 255) / 256)), dim3(256), 0, stream>>>(
        in, o, K, N, total);
  };
  auto gemm = [&](const __bf16* A, long lda, const __bf16* Bt, long ldb,
                  const float* bias, const float* Cin, float* Cout, __bf16* Cb,
                  long ldc, int M, int N, int K, float scale, int relu,
                  int nb, int HH, long sAb, long sAh, long sBb, long sBh,
                  long sCb, long sCh) {
    dim3 grid((unsigned)(N / 64), (unsigned)(M / 256), (unsigned)nb);
    dim3 blk(256);
#define GEMM_ARGS A, lda, Bt, ldb, bias, Cin, Cout, Cb, ldc, M, N, K, scale, \
                  HH, sAb, sAh, sBb, sBh, sCb, sCh
    if (Cb && !bias)                 // QKV / ctx projections, P@V
      gemm_wmma<false, false, false, false, true><<<grid, blk, 0, stream>>>(GEMM_ARGS);
    else if (Cb && bias && relu)     // FFN layer 1
      gemm_wmma<true, false, true, false, true><<<grid, blk, 0, stream>>>(GEMM_ARGS);
    else if (Cout && !bias && !Cin)  // attention scores
      gemm_wmma<false, false, false, true, false><<<grid, blk, 0, stream>>>(GEMM_ARGS);
    else if (Cout && !bias && Cin)   // attention out-proj + residual
      gemm_wmma<false, true, false, true, false><<<grid, blk, 0, stream>>>(GEMM_ARGS);
    else if (Cout && bias && Cin)    // FFN layer 2 + residual
      gemm_wmma<true, true, false, true, false><<<grid, blk, 0, stream>>>(GEMM_ARGS);
    else                             // final vocab projection (bias, f32 out)
      gemm_wmma<true, false, false, true, false><<<grid, blk, 0, stream>>>(GEMM_ARGS);
#undef GEMM_ARGS
  };
  auto ln = [&](const float* X, const float* g, const float* b, __bf16* Z) {
    layernorm_bf16<<<dim3(kNT / 8), dim3(256), 0, stream>>>(X, g, b, Z, kNT);
  };
  auto smax = [&](int causal) {
    softmax1024<<<dim3(kB * kH * kS / 8), dim3(256), 0, stream>>>(
        scores, probs, causal, kB * kH * kS);
  };
  // Multi-head attention: zq queries, zkv keys/values, W = 4 stacked [N,K]
  // transposed D x D matrices (WqT, WkT, WvT, WoT); accumulates into xres.
  auto attention = [&](const __bf16* zq, const __bf16* zkv, const __bf16* W,
                       float* xres, int causal) {
    gemm(zq,  kD, W + 0L * kD * kD, kD, nullptr, nullptr, nullptr, qb,
         kD, kNT, kD, kD, 1.0f, 0, 1, 1, 0, 0, 0, 0, 0, 0);
    gemm(zkv, kD, W + 1L * kD * kD, kD, nullptr, nullptr, nullptr, kbuf,
         kD, kNT, kD, kD, 1.0f, 0, 1, 1, 0, 0, 0, 0, 0, 0);
    gemm(zkv, kD, W + 2L * kD * kD, kD, nullptr, nullptr, nullptr, vbuf,
         kD, kNT, kD, kD, 1.0f, 0, 1, 1, 0, 0, 0, 0, 0, 0);
    transpose_v<<<dim3(kB * kH * kHD * kS / 256), dim3(256), 0, stream>>>(vbuf, vT);
    // scores[b,h] = Q[b,:,h] @ K[b,:,h]^T / 8  (kbuf rows are already [n,k])
    gemm(qb, kD, kbuf, kD, nullptr, nullptr, scores, nullptr,
         kS, kS, kS, kHD, 0.125f, 0,
         kB * kH, kH,
         (long)kS * kD, (long)kHD,
         (long)kS * kD, (long)kHD,
         (long)kH * kS * kS, (long)kS * kS);
    smax(causal);
    // ctx[b,:,h] = P[b,h] @ V[b,:,h]   (vT is [b*H+h][64][S])
    gemm(probs, kS, vT, kS, nullptr, nullptr, nullptr, ctx,
         kD, kS, kHD, kS, 1.0f, 0,
         kB * kH, kH,
         (long)kH * kS * kS, (long)kS * kS,
         (long)kH * kHD * kS, (long)kHD * kS,
         (long)kS * kD, (long)kHD);
    // x += ctx @ Wo
    gemm(ctx, kD, W + 3L * kD * kD, kD, nullptr, xres, xres, nullptr,
         kD, kNT, kD, kD, 1.0f, 0, 1, 1, 0, 0, 0, 0, 0, 0);
  };
  auto ffn = [&](const __bf16* w1T, const float* b1, const __bf16* w2T,
                 const float* b2, float* xres) {
    gemm(zb, kD, w1T, kD, b1, nullptr, nullptr, ffh,
         kFF, kNT, kFF, kD, 1.0f, 1, 1, 1, 0, 0, 0, 0, 0, 0);
    gemm(ffh, kFF, w2T, kFF, b2, xres, xres, nullptr,
         kD, kNT, kD, kFF, 1.0f, 0, 1, 1, 0, 0, 0, 0, 0, 0);
  };

  // ---- convert + transpose all GEMM weights to bf16 [N,K] ----
  cvtT(enc_att_w, w_enc_att, kL * 4, kD, kD);
  cvtT(enc_ff_w1, w_enc_f1,  kL, kD, kFF);
  cvtT(enc_ff_w2, w_enc_f2,  kL, kFF, kD);
  cvtT(dec_att_w, w_dec_att, kL * 2 * 4, kD, kD);
  cvtT(dec_ff_w1, w_dec_f1,  kL, kD, kFF);
  cvtT(dec_ff_w2, w_dec_f2,  kL, kFF, kD);
  cvtT(proj_w,    w_proj,    1, kD, kV);

  // ---- embeddings + positional encoding ----
  embed_pe<<<dim3(kNT * kD / 256), dim3(256), 0, stream>>>(src_emb, src, x_enc, kNT);
  embed_pe<<<dim3(kNT * kD / 256), dim3(256), 0, stream>>>(tgt_emb, tgt, x_dec, kNT);

  // ---- encoder ----
  for (int i = 0; i < kL; ++i) {
    const __bf16* W = w_enc_att + (long)i * 4 * kD * kD;
    ln(x_enc, enc_norm_g + (i * 2 + 0) * kD, enc_norm_b + (i * 2 + 0) * kD, zb);
    attention(zb, zb, W, x_enc, /*causal=*/0);
    ln(x_enc, enc_norm_g + (i * 2 + 1) * kD, enc_norm_b + (i * 2 + 1) * kD, zb);
    ffn(w_enc_f1 + (long)i * kFF * kD, enc_ff_b1 + i * kFF,
        w_enc_f2 + (long)i * kD * kFF, enc_ff_b2 + i * kD, x_enc);
  }
  ln(x_enc, enc_fin_g, enc_fin_b, enc_b);

  // ---- decoder ----
  for (int i = 0; i < kL; ++i) {
    const __bf16* Wself  = w_dec_att + ((long)i * 2 + 0) * 4 * kD * kD;
    const __bf16* Wcross = w_dec_att + ((long)i * 2 + 1) * 4 * kD * kD;
    ln(x_dec, dec_norm_g + (i * 3 + 0) * kD, dec_norm_b + (i * 3 + 0) * kD, zb);
    attention(zb, zb, Wself, x_dec, /*causal=*/1);
    ln(x_dec, dec_norm_g + (i * 3 + 1) * kD, dec_norm_b + (i * 3 + 1) * kD, zb);
    attention(zb, enc_b, Wcross, x_dec, /*causal=*/0);
    ln(x_dec, dec_norm_g + (i * 3 + 2) * kD, dec_norm_b + (i * 3 + 2) * kD, zb);
    ffn(w_dec_f1 + (long)i * kFF * kD, dec_ff_b1 + i * kFF,
        w_dec_f2 + (long)i * kD * kFF, dec_ff_b2 + i * kD, x_dec);
  }
  ln(x_dec, dec_fin_g, dec_fin_b, zb);

  // ---- logits = dec @ proj_w + proj_b ----
  gemm(zb, kD, w_proj, kD, proj_b, nullptr, out, nullptr,
       kV, kNT, kV, kD, 1.0f, 0, 1, 1, 0, 0, 0, 0, 0, 0);
}
